// Q_ShareScale_VGG9_57964878626993
// MI455X (gfx1250) — 8-bit matrix operand order repaired and checked by reading, compiled, not run on hardware
//
#include <hip/hip_runtime.h>
#include <stdint.h>

typedef int v8i __attribute__((ext_vector_type(8)));

#define BATCH 128
#define TSTEPS 4
#define NCLS 10
#define M32 (BATCH * 32 * 32)
#define M16 (BATCH * 16 * 16)
#define M8  (BATCH * 8 * 8)

__global__ __launch_bounds__(256) void conv1_kernel(const float* __restrict__ inp,
                                                    const float* __restrict__ w1,
                                                    float* __restrict__ stat) {
    int i = blockIdx.x * 256 + threadIdx.x;
    if (i >= M32 * 64) return;
    int oc = i & 63;
    int m = i >> 6;
    int x = m & 31, y = (m >> 5) & 31, b = m >> 10;
    float acc = 0.f;
#pragma unroll
    for (int c = 0; c < 3; ++c) {
#pragma unroll
        for (int dy = 0; dy < 3; ++dy) {
            int iy = y + dy - 1;
            if (iy < 0 || iy >= 32) continue;
#pragma unroll
            for (int dx = 0; dx < 3; ++dx) {
                int ix = x + dx - 1;
                if (ix < 0 || ix >= 32) continue;
                acc += inp[((b * 3 + c) * 32 + iy) * 32 + ix] *
                       w1[((oc * 3 + c) * 3 + dy) * 3 + dx];
            }
        }
    }
    stat[(size_t)m * 64 + oc] = acc;
}

__global__ __launch_bounds__(256) void absmax_kernel(const float* __restrict__ w, int n,
                                                     float* __restrict__ out) {
    __shared__ float red[256];
    float m = 0.f;
    for (int i = threadIdx.x; i < n; i += 256) m = fmaxf(m, fabsf(w[i]));
    red[threadIdx.x] = m;
    __syncthreads();
    for (int s = 128; s > 0; s >>= 1) {
        if (threadIdx.x < s) red[threadIdx.x] = fmaxf(red[threadIdx.x], red[threadIdx.x + s]);
        __syncthreads();
    }
    if (threadIdx.x == 0) out[0] = red[0];
}

__global__ __launch_bounds__(256) void pack_kernel(const float* __restrict__ w,
                                                   const float* __restrict__ scale_p,
                                                   int8_t* __restrict__ dst,
                                                   int cin, int nkt, int total) {
    int i = blockIdx.x * 256 + threadIdx.x;
    if (i >= total) return;
    float s = scale_p[0];
    int byte = i & 31;
    int lane = (i >> 5) & 31;
    int blk = i >> 10;
    int kt = blk % nkt;
    int ntile = blk / nkt;
    int N = ntile * 16 + (lane & 15);
    int half = lane >> 4;
    int K = kt * 64 + (byte >> 3) * 16 + half * 8 + (byte & 7);
    int e = K / cin, c = K % cin;
    float wv = w[(size_t)(N * cin + c) * 9 + e];
    float q = fminf(1.f, fmaxf(-1.f, rintf(wv / s)));
    dst[i] = (int8_t)(int)q;
}

__global__ __launch_bounds__(256) void lif1_kernel(const float* __restrict__ stat,
                                                   float* __restrict__ u,
                                                   uint8_t* __restrict__ s1, int n) {
    int i = blockIdx.x * 256 + threadIdx.x;
    if (i >= n) return;
    float uu = 0.5f * u[i] + stat[i];
    int sp = (uu >= 1.0f) ? 1 : 0;
    u[i] = uu - (float)sp;
    int c = i & 63, m = i >> 6;
    int x = m & 31, y = (m >> 5) & 31, b = m >> 10;
    s1[((size_t)(b * 34 + y + 1) * 34 + x + 1) * 64 + c] = (uint8_t)sp;
}

template <int CIN, int COUT, int H, int W>
__global__ __launch_bounds__(256) void conv_lif_wmma(const uint8_t* __restrict__ sin,
                                                     const int8_t* __restrict__ wpk,
                                                     const float* __restrict__ scale_p,
                                                     int8_t* __restrict__ u,
                                                     uint8_t* __restrict__ sout) {
    constexpr int NKT = CIN * 9 / 64;
    constexpr int KTE = CIN / 64;
    constexpr int NG = COUT / 64;
    constexpr int HP = H + 2, WP = W + 2;

    __shared__ alignas(16) int8_t bsm[2][4096];

    const int tid = threadIdx.x;
    const int wid = tid >> 5;
    const int lane = tid & 31;
    const int ng = blockIdx.x % NG;
    const int mb = blockIdx.x / NG;
    const int half = lane >> 4;
    const int l15 = lane & 15;

    const int mt0 = (mb * 8 + wid) * 2;
    const int mrow0 = mt0 * 16 + l15;
    const int mrow1 = mrow0 + 16;
    const int x0 = mrow0 % W, y0 = (mrow0 / W) % H, b0 = mrow0 / (W * H);
    const int x1 = mrow1 % W, y1 = (mrow1 / W) % H, b1 = mrow1 / (W * H);

    const size_t cp_src_base = ((size_t)(ng * 4 + (tid >> 6)) * NKT) << 10;
    const int cp_rem = (tid & 63) << 4;

    auto issue_copy = [&](int kt, int buf) {
        uint64_t g = (uint64_t)(uintptr_t)(wpk + cp_src_base + ((size_t)kt << 10) + cp_rem);
        unsigned loff = (unsigned)(unsigned long long)
            (__attribute__((address_space(3))) int8_t*)(&bsm[buf][tid << 4]);
        asm volatile("global_load_async_to_lds_b128 %0, %1, off"
                     :: "v"(loff), "v"(g) : "memory");
    };

    issue_copy(0, 0);

    v8i acc0[4] = {v8i{}, v8i{}, v8i{}, v8i{}};
    v8i acc1[4] = {v8i{}, v8i{}, v8i{}, v8i{}};

    for (int kt = 0; kt < NKT; ++kt) {
        const int e = kt / KTE;
        const int c0 = (kt % KTE) * 64;
        const int dy = e / 3, dx = e % 3;

        const uint8_t* ap0 =
            sin + (size_t)((b0 * HP + y0 + dy) * WP + x0 + dx) * CIN + c0 + half * 8;
        const uint8_t* ap1 =
            sin + (size_t)((b1 * HP + y1 + dy) * WP + x1 + dx) * CIN + c0 + half * 8;
        v8i A0, A1;
        {
            const uint2 p0 = *(const uint2*)(ap0);
            const uint2 p1 = *(const uint2*)(ap0 + 16);
            const uint2 p2 = *(const uint2*)(ap0 + 32);
            const uint2 p3 = *(const uint2*)(ap0 + 48);
            A0[0] = (int)p0.x; A0[1] = (int)p0.y; A0[2] = (int)p1.x; A0[3] = (int)p1.y;
            A0[4] = (int)p2.x; A0[5] = (int)p2.y; A0[6] = (int)p3.x; A0[7] = (int)p3.y;
        }
        {
            const uint2 p0 = *(const uint2*)(ap1);
            const uint2 p1 = *(const uint2*)(ap1 + 16);
            const uint2 p2 = *(const uint2*)(ap1 + 32);
            const uint2 p3 = *(const uint2*)(ap1 + 48);
            A1[0] = (int)p0.x; A1[1] = (int)p0.y; A1[2] = (int)p1.x; A1[3] = (int)p1.y;
            A1[4] = (int)p2.x; A1[5] = (int)p2.y; A1[6] = (int)p3.x; A1[7] = (int)p3.y;
        }

        asm volatile("s_wait_asynccnt 0x0" ::: "memory");
        __syncthreads();
        if (kt + 1 < NKT) issue_copy(kt + 1, (kt + 1) & 1);

        const int8_t* bb = &bsm[kt & 1][lane << 5];
#pragma unroll
        for (int nt = 0; nt < 4; ++nt) {
            const int4 q0 = *(const int4*)(bb + nt * 1024);
            const int4 q1 = *(const int4*)(bb + nt * 1024 + 16);
            v8i Bf;
            Bf[0] = q0.x; Bf[1] = q0.y; Bf[2] = q0.z; Bf[3] = q0.w;
            Bf[4] = q1.x; Bf[5] = q1.y; Bf[6] = q1.z; Bf[7] = q1.w;
            acc0[nt] = __builtin_amdgcn_wmma_i32_16x16x64_iu8(
                false, A0, true, Bf, acc0[nt], false, false);
            acc1[nt] = __builtin_amdgcn_wmma_i32_16x16x64_iu8(
                false, A1, true, Bf, acc1[nt], false, false);
        }
    }

    const float wsc = scale_p[0];
#pragma unroll
    for (int tile = 0; tile < 2; ++tile) {
        const v8i* acc = tile ? acc1 : acc0;
        const int mbase = (mt0 + tile) * 16;
#pragma unroll
        for (int r = 0; r < 8; ++r) {
            const int mo = mbase + r + 8 * half;
            const int xo = mo % W, yo = (mo / W) % H, bo = mo / (W * H);
            const size_t prow = ((size_t)(bo * HP + yo + 1) * WP + xo + 1) * COUT;
#pragma unroll
            for (int nt = 0; nt < 4; ++nt) {
                const int no = ng * 64 + nt * 16 + l15;
                const size_t uidx = (size_t)mo * COUT + no;
                float xv = (float)acc[nt][r] * wsc;
                float f = 0.5f * (float)u[uidx] + xv;
                float q = fminf(1.f, fmaxf(-1.f, rintf(f)));
                int qi = (int)q;
                int sp = (qi >= 1) ? 1 : 0;
                u[uidx] = (int8_t)(qi - sp);
                sout[prow + no] = (uint8_t)sp;
            }
        }
    }
}

template <int C>
__global__ __launch_bounds__(256) void pool_kernel(const uint8_t* __restrict__ in,
                                                   uint8_t* __restrict__ out,
                                                   int Ho, int Wo, int total) {
    int i = blockIdx.x * 256 + threadIdx.x;
    if (i >= total) return;
    int c = i % C;
    int m = i / C;
    int xo = m % Wo, yo = (m / Wo) % Ho, b = m / (Wo * Ho);
    int WiP = Wo * 2 + 2, HiP = Ho * 2 + 2, WoP = Wo + 2, HoP = Ho + 2;
    const uint8_t* p = in + ((size_t)(b * HiP + 2 * yo + 1) * WiP + 2 * xo + 1) * C + c;
    uint8_t v0 = p[0], v1 = p[C];
    uint8_t v2 = p[(size_t)WiP * C], v3 = p[(size_t)WiP * C + C];
    uint8_t v = v0 > v1 ? v0 : v1;
    uint8_t w = v2 > v3 ? v2 : v3;
    out[((size_t)(b * HoP + yo + 1) * WoP + xo + 1) * C + c] = v > w ? v : w;
}

__global__ __launch_bounds__(256) void avg_kernel(const uint8_t* __restrict__ s,
                                                  float* __restrict__ feat) {
    int i = blockIdx.x * 256 + threadIdx.x;
    if (i >= BATCH * 256) return;
    int c = i & 255, b = i >> 8;
    int sum = 0;
#pragma unroll
    for (int y = 0; y < 8; ++y)
#pragma unroll
        for (int x = 0; x < 8; ++x)
            sum += s[((size_t)(b * 10 + y + 1) * 10 + x + 1) * 256 + c];
    feat[i] = (float)sum * (1.0f / 64.0f);
}

__global__ __launch_bounds__(256) void fc_kernel(const float* __restrict__ feat,
                                                 const float* __restrict__ fw,
                                                 const float* __restrict__ fb,
                                                 float* __restrict__ out) {
    int i = blockIdx.x * 256 + threadIdx.x;
    if (i >= BATCH * NCLS) return;
    int cls = i % NCLS, b = i / NCLS;
    float acc = fb[cls];
    for (int c = 0; c < 256; ++c) acc += feat[b * 256 + c] * fw[cls * 256 + c];
    out[i] = acc;
}

extern "C" void kernel_launch(void* const* d_in, const int* in_sizes, int n_in,
                              void* d_out, int out_size, void* d_ws, size_t ws_size,
                              hipStream_t stream) {
    (void)in_sizes; (void)n_in; (void)out_size; (void)ws_size;
    const float* inp = (const float*)d_in[0];
    const float* w1  = (const float*)d_in[1];
    const float* w2  = (const float*)d_in[2];
    const float* w3  = (const float*)d_in[3];
    const float* w4  = (const float*)d_in[4];
    const float* w5  = (const float*)d_in[5];
    const float* w6  = (const float*)d_in[6];
    const float* w7  = (const float*)d_in[7];
    const float* fcw = (const float*)d_in[8];
    const float* fcb = (const float*)d_in[9];
    float* out = (float*)d_out;

    const size_t S1 = (size_t)BATCH * 34 * 34 * 64;
    const size_t S2P = (size_t)BATCH * 18 * 18 * 64;
    const size_t S3 = (size_t)BATCH * 18 * 18 * 128;
    const size_t S4P = (size_t)BATCH * 10 * 10 * 128;
    const size_t S5 = (size_t)BATCH * 10 * 10 * 256;

    uint8_t* base = (uint8_t*)d_ws;
    size_t off = 0;
    auto alloc = [&](size_t bytes) -> void* {
        void* p = base + off;
        off += (bytes + 255) & ~(size_t)255;
        return p;
    };
    float*   stat = (float*)alloc((size_t)M32 * 64 * 4);
    float*   u1   = (float*)alloc((size_t)M32 * 64 * 4);
    int8_t*  u2   = (int8_t*)alloc((size_t)M32 * 64);
    int8_t*  u3   = (int8_t*)alloc((size_t)M16 * 128);
    int8_t*  u4   = (int8_t*)alloc((size_t)M16 * 128);
    int8_t*  u5   = (int8_t*)alloc((size_t)M8 * 256);
    int8_t*  u6   = (int8_t*)alloc((size_t)M8 * 256);
    int8_t*  u7   = (int8_t*)alloc((size_t)M8 * 256);
    uint8_t* s1   = (uint8_t*)alloc(S1);
    uint8_t* s2   = (uint8_t*)alloc(S1);
    uint8_t* s2p  = (uint8_t*)alloc(S2P);
    uint8_t* s3   = (uint8_t*)alloc(S3);
    uint8_t* s4   = (uint8_t*)alloc(S3);
    uint8_t* s4p  = (uint8_t*)alloc(S4P);
    uint8_t* s5   = (uint8_t*)alloc(S5);
    uint8_t* s6   = (uint8_t*)alloc(S5);
    uint8_t* s7   = (uint8_t*)alloc(S5);
    int8_t*  qw2  = (int8_t*)alloc(576 * 64);
    int8_t*  qw3  = (int8_t*)alloc(576 * 128);
    int8_t*  qw4  = (int8_t*)alloc(1152 * 128);
    int8_t*  qw5  = (int8_t*)alloc(1152 * 256);
    int8_t*  qw6  = (int8_t*)alloc(2304 * 256);
    int8_t*  qw7  = (int8_t*)alloc(2304 * 256);
    float*   scl  = (float*)alloc(6 * 4);
    float*   feat = (float*)alloc((size_t)BATCH * 256 * 4);

    hipMemsetAsync(u1, 0, (size_t)M32 * 64 * 4, stream);
    hipMemsetAsync(u2, 0, (size_t)M32 * 64, stream);
    hipMemsetAsync(u3, 0, (size_t)M16 * 128, stream);
    hipMemsetAsync(u4, 0, (size_t)M16 * 128, stream);
    hipMemsetAsync(u5, 0, (size_t)M8 * 256, stream);
    hipMemsetAsync(u6, 0, (size_t)M8 * 256, stream);
    hipMemsetAsync(u7, 0, (size_t)M8 * 256, stream);
    hipMemsetAsync(s1, 0, S1, stream);
    hipMemsetAsync(s2p, 0, S2P, stream);
    hipMemsetAsync(s3, 0, S3, stream);
    hipMemsetAsync(s4p, 0, S4P, stream);
    hipMemsetAsync(s5, 0, S5, stream);
    hipMemsetAsync(s6, 0, S5, stream);

    conv1_kernel<<<(M32 * 64 + 255) / 256, 256, 0, stream>>>(inp, w1, stat);

    struct { const float* w; int cin, nkt; int8_t* dst; } L[6] = {
        {w2,  64,  9, qw2}, {w3,  64,  9, qw3}, {w4, 128, 18, qw4},
        {w5, 128, 18, qw5}, {w6, 256, 36, qw6}, {w7, 256, 36, qw7},
    };
    const int couts[6] = {64, 128, 128, 256, 256, 256};
    for (int l = 0; l < 6; ++l) {
        int n = L[l].cin * couts[l] * 9;
        absmax_kernel<<<1, 256, 0, stream>>>(L[l].w, n, scl + l);
        pack_kernel<<<(n + 255) / 256, 256, 0, stream>>>(L[l].w, scl + l, L[l].dst,
                                                         L[l].cin, L[l].nkt, n);
    }

    for (int t = 0; t < TSTEPS; ++t) {
        lif1_kernel<<<(M32 * 64 + 255) / 256, 256, 0, stream>>>(stat, u1, s1, M32 * 64);
        conv_lif_wmma<64, 64, 32, 32><<<512, 256, 0, stream>>>(s1, qw2, scl + 0, u2, s2);
        pool_kernel<64><<<(M16 * 64 + 255) / 256, 256, 0, stream>>>(s2, s2p, 16, 16, M16 * 64);
        conv_lif_wmma<64, 128, 16, 16><<<256, 256, 0, stream>>>(s2p, qw3, scl + 1, u3, s3);
        conv_lif_wmma<128, 128, 16, 16><<<256, 256, 0, stream>>>(s3, qw4, scl + 2, u4, s4);
        pool_kernel<128><<<(M8 * 128 + 255) / 256, 256, 0, stream>>>(s4, s4p, 8, 8, M8 * 128);
        conv_lif_wmma<128, 256, 8, 8><<<128, 256, 0, stream>>>(s4p, qw5, scl + 3, u5, s5);
        conv_lif_wmma<256, 256, 8, 8><<<128, 256, 0, stream>>>(s5, qw6, scl + 4, u6, s6);
        conv_lif_wmma<256, 256, 8, 8><<<128, 256, 0, stream>>>(s6, qw7, scl + 5, u7, s7);
        avg_kernel<<<(BATCH * 256 + 255) / 256, 256, 0, stream>>>(s7, feat);
        fc_kernel<<<(BATCH * NCLS + 255) / 256, 256, 0, stream>>>(feat, fcw, fcb,
                                                                  out + t * BATCH * NCLS);
    }
}
